// MultiHeadAttn_36550171689428
// MI455X (gfx1250) — matrix operand order repaired and checked by reading, not compiled, not run on hardware
//
#include <hip/hip_runtime.h>
#include <math.h>

// ---------------------------------------------------------------------------
// MI455X (gfx1250) multi-head attention, bf16 WMMA, flash-style streaming.
// Shapes (fixed by reference): B=2, LQ=LK=2048, D_MODEL=1024, H=16, dh=64.
// ---------------------------------------------------------------------------

#define D_MODEL 1024
#define NH      16
#define DH      64
#define SEQ     2048
#define BATCH   2
#define SCALE_INV 0.125f          // 1/sqrt(64)
#define MASK_SENT (-1.0e30f)

typedef __attribute__((ext_vector_type(16))) __bf16 v16bf;
typedef __attribute__((ext_vector_type(8)))  float  v8f;

__device__ __forceinline__ v8f wmma_bf16(v16bf a, v16bf b, v8f c) {
  // D = A(16x32 bf16) * B(32x16 bf16) + C(16x16 f32)
  return __builtin_amdgcn_wmma_f32_16x16x32_bf16(
      false, a, false, b, (short)0, c, false, false);
}

// A-operand (16x32 bf16): lane (c,hi) holds row c; k = 16*hi + [0..15]
__device__ __forceinline__ v16bf load_A_bf(const __bf16* __restrict__ row, int k0, int hi) {
  v16bf a;
#pragma unroll
  for (int i = 0; i < 8; ++i) a[i]     = row[k0 + 16*hi + i];
#pragma unroll
  for (int i = 0; i < 8; ++i) a[8 + i] = row[k0 + 16*hi + 8 + i];
  return a;
}

// B-operand (32x16 bf16): lane (c,hi) holds column c; k = 16*hi + [0..15]
__device__ __forceinline__ v16bf load_B_bf(const __bf16* __restrict__ row, int k0, int hi) {
  v16bf b;
#pragma unroll
  for (int i = 0; i < 16; ++i) b[i] = row[k0 + 16*hi + i];
  return b;
}

// ---------------------------------------------------------------------------
// Kernel 0a: f32 -> bf16 streaming conversion (weights, activations).
// ---------------------------------------------------------------------------
__global__ void cvt_kernel(const float* __restrict__ src,
                           __bf16* __restrict__ dst, int n) {
  int i = blockIdx.x * blockDim.x + threadIdx.x;
  if (i < n) dst[i] = (__bf16)src[i];
}

// ---------------------------------------------------------------------------
// Kernel 1: projection  out[m][n] = sum_d X[m][d] * W[n][d] + bias[n]
// One wave computes a 16x64 tile (4 accumulators): A fragment loaded once per
// K-step and fed to 4 WMMAs.  Inner loop is pure bf16 b128 loads + WMMA.
// Output layouts:
//   outN : [b][h][l][d]  (row-contiguous over d)   -- B-operand of Q*K^T
//   outT : [b][h][d][l]  (row-contiguous over l)   -- B-operand of P*K, P*V
// ---------------------------------------------------------------------------
__global__ __launch_bounds__(256)
void proj_kernel(const __bf16* __restrict__ Xb,
                 const __bf16* __restrict__ W,
                 const float* __restrict__ bias,
                 __bf16* __restrict__ outN,
                 __bf16* __restrict__ outT,
                 int writeN, int writeT) {
  const int wave = threadIdx.x >> 5;
  const int lane = threadIdx.x & 31;
  const int c  = lane & 15;
  const int hi = lane >> 4;

  const int m_tile = blockIdx.x;                 // 16 rows of B*L = 4096
  const int n_grp  = blockIdx.y * 8 + wave;      // 0..15, 64 cols each == head
  const int n_base = n_grp * 64;

  const __bf16* xrow = Xb + (size_t)(m_tile * 16 + c) * D_MODEL;   // A row
  const __bf16* wrow[4];
#pragma unroll
  for (int t = 0; t < 4; ++t)
    wrow[t] = W + (size_t)(n_base + 16 * t + c) * D_MODEL;         // B column c

  v8f acc[4];
#pragma unroll
  for (int t = 0; t < 4; ++t) { v8f z = {}; acc[t] = z; }

#pragma unroll 2
  for (int kk = 0; kk < D_MODEL; kk += 32) {
    const v16bf a = load_A_bf(xrow, kk, hi);
#pragma unroll
    for (int t = 0; t < 4; ++t)
      acc[t] = wmma_bf16(a, load_B_bf(wrow[t], kk, hi), acc[t]);
  }

  const int h = n_grp;                         // n_base is a multiple of 64
#pragma unroll
  for (int t = 0; t < 4; ++t) {
    const int d  = 16 * t + c;
    const float bn = bias[n_base + d];
#pragma unroll
    for (int r = 0; r < 8; ++r) {
      const int M     = r + 8 * hi;
      const int m_abs = m_tile * 16 + M;
      const int bidx  = m_abs >> 11;           // /2048
      const int l     = m_abs & (SEQ - 1);
      const __bf16 vbf = (__bf16)(acc[t][r] + bn);
      const size_t bh = (size_t)bidx * NH + h;
      if (writeN) outN[(bh * SEQ + l) * DH + d] = vbf;
      if (writeT) outT[(bh * DH + d) * SEQ + l] = vbf;
    }
  }
}

// ---------------------------------------------------------------------------
// Kernel 2: flash attention producing BOTH  k_out = P*Kh  and  v_out = P*Vh.
// Block = (b, 16-row q tile); 16 waves = 16 heads share the LDS mask chunk.
// Finite -1e30 mask sentinel reproduces the reference's empty-row -> uniform
// softmax exactly (all-masked rows: exp(0)=1 everywhere; transient uniform
// contributions are annihilated by the online-softmax correction factor).
// ---------------------------------------------------------------------------
__global__ __launch_bounds__(512, 1)
void flash_kernel(const __bf16* __restrict__ Qh,   // [bh][l][d]
                  const __bf16* __restrict__ KhN,  // [bh][l][d]
                  const __bf16* __restrict__ KhT,  // [bh][d][l]
                  const __bf16* __restrict__ VhT,  // [bh][d][l]
                  const int*    __restrict__ mask, // [b][lq][lk]
                  float* __restrict__ k_out,
                  float* __restrict__ v_out) {
  __shared__ int    smask[16][256];     // mask chunk: 16 q-rows x 256 k
  __shared__ __bf16 pscr[16][16 * 32];  // per-wave P(16x32) C->A relayout

  const int wave = threadIdx.x >> 5;    // == head
  const int lane = threadIdx.x & 31;
  const int c  = lane & 15;
  const int hi = lane >> 4;

  const int bidx = blockIdx.x >> 7;           // 128 q-tiles per batch
  const int q0   = (blockIdx.x & 127) * 16;
  const size_t headoff = ((size_t)bidx * NH + wave) * SEQ * DH; // same for N & T

  // Preload this wave's Q tile as two A fragments (d 0..31, d 32..63).
  const __bf16* qrow = Qh + headoff + (size_t)(q0 + c) * DH;
  const v16bf aq0 = load_A_bf(qrow, 0, hi);
  const v16bf aq1 = load_A_bf(qrow, 32, hi);

  v8f ov[4], ok[4];
#pragma unroll
  for (int t = 0; t < 4; ++t) { v8f z = {}; ov[t] = z; ok[t] = z; }
  float mrow[8], lrow[8];
#pragma unroll
  for (int r = 0; r < 8; ++r) { mrow[r] = -3.0e38f; lrow[r] = 0.0f; }

  const int* gmask = mask + ((size_t)bidx * SEQ + q0) * SEQ;

  for (int kc0 = 0; kc0 < SEQ; kc0 += 256) {
    __syncthreads();  // previous chunk fully consumed before overwrite
#pragma unroll
    for (int it = 0; it < 8; ++it) {
      const int idx = threadIdx.x + it * 512;      // 16*256 = 4096 ints
      smask[idx >> 8][idx & 255] = gmask[(size_t)(idx >> 8) * SEQ + kc0 + (idx & 255)];
    }
    if (kc0 + 256 < SEQ && threadIdx.x < 16)        // prefetch next mask chunk
      __builtin_prefetch(&gmask[(size_t)threadIdx.x * SEQ + kc0 + 256], 0, 1);
    __syncthreads();

    for (int n0 = 0; n0 < 256; n0 += 32) {
      const int kg = kc0 + n0;                     // global k start of this step
      // ---- S = Q * K^T for two 16-col tiles (k cols kg..kg+31) ----
      const __bf16* krow_lo = KhN + headoff + (size_t)(kg + c) * DH;
      const __bf16* krow_hi = KhN + headoff + (size_t)(kg + 16 + c) * DH;
      v8f slo = {}, shi = {};
      slo = wmma_bf16(aq0, load_B_bf(krow_lo, 0, hi), slo);
      slo = wmma_bf16(aq1, load_B_bf(krow_lo, 32, hi), slo);
      shi = wmma_bf16(aq0, load_B_bf(krow_hi, 0, hi), shi);
      shi = wmma_bf16(aq1, load_B_bf(krow_hi, 32, hi), shi);

      // ---- scale, mask, online softmax (rows live in (r, 16-lane half)) ----
#pragma unroll
      for (int r = 0; r < 8; ++r) {
        const int M = r + 8 * hi;
        float sl = slo[r] * SCALE_INV;
        float sh = shi[r] * SCALE_INV;
        if (smask[M][n0 + c] == 0)      sl = MASK_SENT;
        if (smask[M][n0 + 16 + c] == 0) sh = MASK_SENT;

        float t = fmaxf(sl, sh);
#pragma unroll
        for (int s = 1; s < 16; s <<= 1) t = fmaxf(t, __shfl_xor(t, s, 32));
        const float mnew  = fmaxf(mrow[r], t);
        const float corrf = __expf(mrow[r] - mnew);
        mrow[r] = mnew;

        const float pl = __expf(sl - mnew);
        const float ph = __expf(sh - mnew);
        float psum = pl + ph;
#pragma unroll
        for (int s = 1; s < 16; s <<= 1) psum += __shfl_xor(psum, s, 32);
        lrow[r] = lrow[r] * corrf + psum;

#pragma unroll
        for (int t4 = 0; t4 < 4; ++t4) { ov[t4][r] *= corrf; ok[t4][r] *= corrf; }

        pscr[wave][M * 32 + c]      = (__bf16)pl;   // C-layout -> LDS
        pscr[wave][M * 32 + 16 + c] = (__bf16)ph;
      }
      asm volatile("s_wait_dscnt 0" ::: "memory");  // wave-local LDS RAW

      // ---- P(16x32) back as A fragment; accumulate P*V and P*K ----
      const v16bf ap = load_A_bf(&pscr[wave][c * 32], 0, hi);
#pragma unroll
      for (int t4 = 0; t4 < 4; ++t4) {
        const __bf16* vrow = VhT + headoff + (size_t)(16 * t4 + c) * SEQ;
        const __bf16* krow = KhT + headoff + (size_t)(16 * t4 + c) * SEQ;
        ov[t4] = wmma_bf16(ap, load_B_bf(vrow, kg, hi), ov[t4]);
        ok[t4] = wmma_bf16(ap, load_B_bf(krow, kg, hi), ok[t4]);
      }
    }
  }

  // ---- normalize and store (outputs are [b][l][h*64+d] f32) ----
#pragma unroll
  for (int r = 0; r < 8; ++r) {
    const float innv = 1.0f / lrow[r];
    const int M = r + 8 * hi;
    const size_t rowo = ((size_t)bidx * SEQ + q0 + M) * D_MODEL + wave * DH;
#pragma unroll
    for (int t4 = 0; t4 < 4; ++t4) {
      const size_t o = rowo + 16 * t4 + c;
      k_out[o] = ok[t4][r] * innv;
      v_out[o] = ov[t4][r] * innv;
    }
  }
}

// ---------------------------------------------------------------------------
// Launch.  Workspace layout (bytes):
//   [0,6M)    Wq/Wk/Wv bf16
//   [6M,14M)  Xb (bf16 activations, reused serially for q, then k, then v)
//   [14M,22M) Qh   [22M,30M) KhN   [30M,38M) KhT   [38M,46M) VhT
// ---------------------------------------------------------------------------
extern "C" void kernel_launch(void* const* d_in, const int* in_sizes, int n_in,
                              void* d_out, int out_size, void* d_ws, size_t ws_size,
                              hipStream_t stream) {
  const float* q    = (const float*)d_in[0];
  const float* k    = (const float*)d_in[1];
  const float* v    = (const float*)d_in[2];
  const int*   mask = (const int*)d_in[3];
  const float* Wq   = (const float*)d_in[4];
  const float* bq   = (const float*)d_in[5];
  const float* Wk   = (const float*)d_in[6];
  const float* bk   = (const float*)d_in[7];
  const float* Wv   = (const float*)d_in[8];
  const float* bv   = (const float*)d_in[9];

  float* k_out = (float*)d_out;
  float* v_out = k_out + (size_t)BATCH * SEQ * D_MODEL;

  char* ws = (char*)d_ws;
  __bf16* Wb  = (__bf16*)(ws);
  __bf16* Xb  = (__bf16*)(ws + (size_t)(6u)  * (1u << 20));
  __bf16* Qh  = (__bf16*)(ws + (size_t)(14u) * (1u << 20));
  __bf16* KhN = (__bf16*)(ws + (size_t)(22u) * (1u << 20));
  __bf16* KhT = (__bf16*)(ws + (size_t)(30u) * (1u << 20));
  __bf16* VhT = (__bf16*)(ws + (size_t)(38u) * (1u << 20));

  const int WN = D_MODEL * D_MODEL;              // 1M elements per weight
  const int XN = BATCH * SEQ * D_MODEL;          // 4M elements per activation
  const int CT = 256;

  // Weights -> bf16 (once).
  cvt_kernel<<<(WN + CT - 1) / CT, CT, 0, stream>>>(Wq, Wb,          WN);
  cvt_kernel<<<(WN + CT - 1) / CT, CT, 0, stream>>>(Wk, Wb + WN,     WN);
  cvt_kernel<<<(WN + CT - 1) / CT, CT, 0, stream>>>(Wv, Wb + 2 * WN, WN);

  dim3 pgrid(256, 2);  // 256 row-tiles x (2 * 8 waves) * 64 cols = 4096x1024

  // q -> Qh
  cvt_kernel<<<(XN + CT - 1) / CT, CT, 0, stream>>>(q, Xb, XN);
  proj_kernel<<<pgrid, 256, 0, stream>>>(Xb, Wb,          bq, Qh,  nullptr, 1, 0);
  // k -> KhN + KhT
  cvt_kernel<<<(XN + CT - 1) / CT, CT, 0, stream>>>(k, Xb, XN);
  proj_kernel<<<pgrid, 256, 0, stream>>>(Xb, Wb + WN,     bk, KhN, KhT,     1, 1);
  // v -> VhT
  cvt_kernel<<<(XN + CT - 1) / CT, CT, 0, stream>>>(v, Xb, XN);
  proj_kernel<<<pgrid, 256, 0, stream>>>(Xb, Wb + 2 * WN, bv, nullptr, VhT, 0, 1);

  flash_kernel<<<BATCH * (SEQ / 16), 512, 0, stream>>>(Qh, KhN, KhT, VhT, mask,
                                                       k_out, v_out);
}
